// MDTA_cross_8993661518346
// MI455X (gfx1250) — compile-verified, matrix operand order checked by reading, not run on hardware
//
#include <hip/hip_runtime.h>
#include <hip/hip_bf16.h>
#include <math.h>

// Problem constants
#define BB   8
#define CC   192
#define C3   576
#define HH   64
#define WW   64
#define HWN  4096
#define HEADS 4
#define DH   48

typedef __attribute__((ext_vector_type(16))) __bf16 v16bf;
typedef __attribute__((ext_vector_type(8)))  __bf16 v8bf;
typedef __attribute__((ext_vector_type(8)))  float  v8f;
typedef __attribute__((ext_vector_type(2)))  float  v2f;

#if __has_builtin(__builtin_amdgcn_wmma_f32_16x16x4_f32)
#define HAVE_WMMA_F32 1
#endif

// Concatenate two contiguous 16-byte bf16 halves into a WMMA A/B operand.
// Half 0 = K offsets (8*hi)+0..7, half 1 = K offsets 16+(8*hi)+0..7 per the
// ISA 16-bit 16x32 operand layout (wave32).
__device__ __forceinline__ v16bf cat16(v8bf lo, v8bf hi8) {
    return __builtin_shufflevector(lo, hi8, 0, 1, 2, 3, 4, 5, 6, 7,
                                   8, 9, 10, 11, 12, 13, 14, 15);
}

// ---------------------------------------------------------------------------
// Kernel 0a: f32 -> bf16 elementwise convert (weights)
// ---------------------------------------------------------------------------
__global__ void cvt_bf16(const float* __restrict__ src, __bf16* __restrict__ dst, int n) {
    int i = blockIdx.x * 256 + threadIdx.x;
    if (i < n) dst[i] = (__bf16)src[i];
}

// ---------------------------------------------------------------------------
// Kernel 0b: transpose-convert  src[b][ic][n] f32  ->  dst[b][n][ic] bf16
// 32x32 LDS tiles; coalesced on both sides. Blocks: b * 6(ic) * 128(n).
// ---------------------------------------------------------------------------
__global__ void transpose_cvt(const float* __restrict__ src, __bf16* __restrict__ dst) {
    __shared__ float tile[32][33];
    int bt  = blockIdx.x;
    int ntile = bt & 127;
    int ict   = (bt >> 7) % 6;
    int b     = bt / (128 * 6);
    int tx = threadIdx.x & 31;
    int ty = threadIdx.x >> 5;  // 8 rows per pass

    const float* sb = src + ((size_t)b * CC + ict * 32) * HWN + ntile * 32;
    for (int r = ty; r < 32; r += 8)
        tile[r][tx] = sb[(size_t)r * HWN + tx];
    __syncthreads();
    __bf16* db = dst + ((size_t)b * HWN + ntile * 32) * CC + ict * 32;
    for (int r = ty; r < 32; r += 8)
        db[(size_t)r * CC + tx] = (__bf16)tile[tx][r];
}

// ---------------------------------------------------------------------------
// Kernel 1: 1x1 conv as GEMM  t[b, oc, n] = sum_ic w[oc,ic] * x[b,ic,n]
// M=576, N=4096, K=192 per batch. A = wbf (ic-contiguous), B = xT (ic-
// contiguous at fixed n): every operand half is one b128 load.
// One wave per 16x16 tile; waves = 8*36*256 = 73728 -> 9216 blocks.
// ---------------------------------------------------------------------------
__global__ void qkv1x1_wmma(const __bf16* __restrict__ xT,
                            const __bf16* __restrict__ wbf,
                            __bf16* __restrict__ tout) {
    int wave = blockIdx.x * 8 + (threadIdx.x >> 5);
    int lane = threadIdx.x & 31;
    int nt  = wave & 255;             // N tile (0..255)
    int oct = (wave >> 8) % 36;       // M tile (0..35)
    int b   = wave / (256 * 36);      // batch
    int m  = lane & 15;
    int hi = lane >> 4;

    const __bf16* arow = wbf + (size_t)(oct * 16 + m) * CC + 8 * hi;
    const __bf16* brow = xT + ((size_t)b * HWN + nt * 16 + m) * CC + 8 * hi;

    v8f c = {};
#pragma unroll
    for (int kk = 0; kk < CC; kk += 32) {
        v8bf alo = *(const v8bf*)(arow + kk);
        v8bf ahi = *(const v8bf*)(arow + kk + 16);
        v8bf blo = *(const v8bf*)(brow + kk);
        v8bf bhi = *(const v8bf*)(brow + kk + 16);
        c = __builtin_amdgcn_wmma_f32_16x16x32_bf16(false, cat16(alo, ahi),
                                                    false, cat16(blo, bhi),
                                                    (short)0, c, false, false);
    }
#pragma unroll
    for (int r = 0; r < 8; ++r) {
        int row = oct * 16 + r + 8 * hi;
        tout[((size_t)b * C3 + row) * HWN + nt * 16 + m] = (__bf16)c[r];
    }
}

// ---------------------------------------------------------------------------
// Kernel 2: depthwise 3x3 (pad 1) + split into q/k/v + row inv-norms for q,k.
// One block per (b, ch) 64x64 plane; halo staged in LDS; block-local
// deterministic reduction for sum-of-squares.
// ---------------------------------------------------------------------------
__global__ void dw3x3_split(const __bf16* __restrict__ tin,
                            const float* __restrict__ wdw,
                            __bf16* __restrict__ qdst,
                            __bf16* __restrict__ kdst,
                            float* __restrict__ vdst,
                            float* __restrict__ invnq,
                            float* __restrict__ invnk) {
    __shared__ float tile[66 * 66];
    __shared__ float red[256];
    int b  = blockIdx.x / C3;
    int ch = blockIdx.x % C3;
    const __bf16* tbase = tin + ((size_t)b * C3 + ch) * HWN;

    for (int t = threadIdx.x; t < 66 * 66; t += 256) {
        int yy = t / 66 - 1, xx = t % 66 - 1;
        float v = 0.f;
        if (yy >= 0 && yy < HH && xx >= 0 && xx < WW)
            v = (float)tbase[yy * WW + xx];
        tile[t] = v;
    }
    float wloc[9];
#pragma unroll
    for (int j = 0; j < 9; ++j) wloc[j] = wdw[ch * 9 + j];
    __syncthreads();

    int part = ch / CC, ch192 = ch % CC;
    size_t obase = ((size_t)b * CC + ch192) * HWN;
    float ssq = 0.f;
    for (int k = 0; k < 16; ++k) {
        int p  = threadIdx.x + k * 256;
        int py = p >> 6, px = p & 63;
        float acc = 0.f;
#pragma unroll
        for (int dy = 0; dy < 3; ++dy)
#pragma unroll
            for (int dx = 0; dx < 3; ++dx)
                acc = fmaf(tile[(py + dy) * 66 + (px + dx)], wloc[dy * 3 + dx], acc);
        if (part == 0)      { qdst[obase + p] = (__bf16)acc; ssq += acc * acc; }
        else if (part == 1) { kdst[obase + p] = (__bf16)acc; ssq += acc * acc; }
        else                { vdst[obase + p] = acc; }
    }
    if (part < 2) {  // block-uniform branch
        __syncthreads();
        red[threadIdx.x] = ssq;
        __syncthreads();
        for (int s = 128; s > 0; s >>= 1) {
            if ((int)threadIdx.x < s) red[threadIdx.x] += red[threadIdx.x + s];
            __syncthreads();
        }
        if (threadIdx.x == 0) {
            float inv = 1.f / fmaxf(sqrtf(red[0]), 1e-12f);
            (part == 0 ? invnq : invnk)[b * CC + ch192] = inv;
        }
    }
}

// ---------------------------------------------------------------------------
// Kernel 3: raw Gram matrices S = q . k^T  (un-normalized) for the 4 pairs
//   pair 0: (qx,kx)=u   1: (qx,ky)=ud   2: (qy,ky)=d   3: (qy,kx)=du
// Per (b,h,pair): M=N=48 (3x3 tiles of 16), K=4096 (n is contiguous in both
// operands -> pure b128 loads). waves = 8*4*4*9 = 1152 -> 144 blocks.
// ---------------------------------------------------------------------------
__global__ void gram_wmma(const __bf16* __restrict__ qx, const __bf16* __restrict__ kx,
                          const __bf16* __restrict__ qy, const __bf16* __restrict__ ky,
                          float* __restrict__ gram) {
    int wave = blockIdx.x * 8 + (threadIdx.x >> 5);
    int lane = threadIdx.x & 31;
    int t9   = wave % 9;
    int pair = (wave / 9) & 3;
    int h    = (wave / 36) & 3;
    int b    = wave / 144;
    int tm = t9 / 3, tn = t9 % 3;
    int m  = lane & 15;
    int hi = lane >> 4;

    const __bf16* qsrc = (pair < 2) ? qx : qy;
    const __bf16* ksrc = (pair == 0 || pair == 3) ? kx : ky;
    const __bf16* arow = qsrc + ((size_t)b * CC + h * DH + tm * 16 + m) * HWN + 8 * hi;
    const __bf16* brow = ksrc + ((size_t)b * CC + h * DH + tn * 16 + m) * HWN + 8 * hi;

    v8f c = {};
#pragma unroll 4
    for (int kk = 0; kk < HWN; kk += 32) {
        v8bf alo = *(const v8bf*)(arow + kk);
        v8bf ahi = *(const v8bf*)(arow + kk + 16);
        v8bf blo = *(const v8bf*)(brow + kk);
        v8bf bhi = *(const v8bf*)(brow + kk + 16);
        c = __builtin_amdgcn_wmma_f32_16x16x32_bf16(false, cat16(alo, ahi),
                                                    false, cat16(blo, bhi),
                                                    (short)0, c, false, false);
    }
    size_t gbase = (((size_t)pair * BB + b) * HEADS + h) * (DH * DH);
#pragma unroll
    for (int r = 0; r < 8; ++r)
        gram[gbase + (size_t)(tm * 16 + r + 8 * hi) * DH + tn * 16 + m] = c[r];
}

// ---------------------------------------------------------------------------
// Kernel 4: fold in 1/|q| * 1/|k| * temperature, then row softmax (48 wide).
// One thread per row; 6144 rows. In-place on gram.
// ---------------------------------------------------------------------------
__global__ void scale_softmax(float* __restrict__ gram,
                              const float* __restrict__ invnqx, const float* __restrict__ invnkx,
                              const float* __restrict__ invnqy, const float* __restrict__ invnky,
                              const float* __restrict__ temp) {
    int gid = blockIdx.x * 256 + threadIdx.x;
    if (gid >= 4 * BB * HEADS * DH) return;
    int cc   = gid % DH;
    int h    = (gid / DH) & 3;
    int b    = (gid / (DH * HEADS)) & 7;
    int pair = gid / (DH * HEADS * BB);
    const float* inq = (pair < 2) ? invnqx : invnqy;
    const float* ink = (pair == 0 || pair == 3) ? invnkx : invnky;
    float qi = inq[b * CC + h * DH + cc];
    float tp = temp[h];
    float* row = gram + (((size_t)pair * BB + b) * HEADS + h) * (DH * DH) + (size_t)cc * DH;

    float s[DH];
    float mx = -3.0e38f;
    for (int d = 0; d < DH; ++d) {
        float v = row[d] * qi * ink[b * CC + h * DH + d] * tp;
        s[d] = v;
        mx = fmaxf(mx, v);
    }
    float sum = 0.f;
    for (int d = 0; d < DH; ++d) { s[d] = __expf(s[d] - mx); sum += s[d]; }
    float inv = 1.f / sum;
    for (int d = 0; d < DH; ++d) row[d] = s[d] * inv;
}

// ---------------------------------------------------------------------------
// Kernel 5: cross = attn_du@vx + attn_ud@vy ; self = attn_u@vx + attn_d@vy
// Per (b,h): M=48, K=48, N=4096, f32 WMMA 16x16x4. One wave computes the
// matching cross AND self tile (v-tile loads shared).
// waves = 8*4*3*256 = 24576 -> 3072 blocks.
// ---------------------------------------------------------------------------
__global__ void attn_apply(const float* __restrict__ attn,
                           const float* __restrict__ vx,
                           const float* __restrict__ vy,
                           float* __restrict__ out) {
#ifdef HAVE_WMMA_F32
    const size_t PAIR_STRIDE = (size_t)BB * HEADS * DH * DH;  // 73728
    int wave = blockIdx.x * 8 + (threadIdx.x >> 5);
    int lane = threadIdx.x & 31;
    int nt = wave & 255;
    int tm = (wave >> 8) % 3;
    int h  = (wave / 768) & 3;
    int b  = wave / 3072;
    int m  = lane & 15;
    int hi = lane >> 4;

    size_t abase = (((size_t)b) * HEADS + h) * (DH * DH) + (size_t)(tm * 16 + m) * DH;
    const float* a_u  = attn + 0 * PAIR_STRIDE + abase;
    const float* a_ud = attn + 1 * PAIR_STRIDE + abase;
    const float* a_d  = attn + 2 * PAIR_STRIDE + abase;
    const float* a_du = attn + 3 * PAIR_STRIDE + abase;
    const float* vxb  = vx + ((size_t)b * CC + h * DH) * HWN + nt * 16 + m;
    const float* vyb  = vy + ((size_t)b * CC + h * DH) * HWN + nt * 16 + m;

    v8f cr = {}, sf = {};
#pragma unroll
    for (int kk = 0; kk < DH; kk += 4) {
        int k0 = kk + 2 * hi;          // f32 16x16x4: K = 2*hi + i
        v2f au, aud, ad, adu, bx, by;
#pragma unroll
        for (int i = 0; i < 2; ++i) {
            au[i]  = a_u[k0 + i];
            aud[i] = a_ud[k0 + i];
            ad[i]  = a_d[k0 + i];
            adu[i] = a_du[k0 + i];
            bx[i]  = vxb[(size_t)(k0 + i) * HWN];
            by[i]  = vyb[(size_t)(k0 + i) * HWN];
        }
        cr = __builtin_amdgcn_wmma_f32_16x16x4_f32(false, adu, false, bx, (short)0, cr, false, false);
        cr = __builtin_amdgcn_wmma_f32_16x16x4_f32(false, aud, false, by, (short)0, cr, false, false);
        sf = __builtin_amdgcn_wmma_f32_16x16x4_f32(false, au,  false, bx, (short)0, sf, false, false);
        sf = __builtin_amdgcn_wmma_f32_16x16x4_f32(false, ad,  false, by, (short)0, sf, false, false);
    }
    const size_t OUT_N = (size_t)BB * CC * HWN;  // 6291456
#pragma unroll
    for (int r = 0; r < 8; ++r) {
        size_t o = ((size_t)b * CC + h * DH + tm * 16 + r + 8 * hi) * HWN + nt * 16 + m;
        out[o]         = cr[r];
        out[OUT_N + o] = sf[r];
    }
#else
    // Exact-FMA fallback (grid-stride)
    const size_t PAIR_STRIDE = (size_t)BB * HEADS * DH * DH;
    const size_t OUT_N = (size_t)BB * CC * HWN;
    for (size_t idx = blockIdx.x * 256ull + threadIdx.x; idx < OUT_N;
         idx += (size_t)gridDim.x * 256ull) {
        int n  = (int)(idx & (HWN - 1));
        int ch = (int)((idx >> 12) % CC);
        int b  = (int)(idx >> 12) / CC;
        int h  = ch / DH, c = ch % DH;
        size_t abase = (((size_t)b) * HEADS + h) * (DH * DH) + (size_t)c * DH;
        size_t vbase = ((size_t)b * CC + h * DH) * HWN + n;
        float crs = 0.f, slf = 0.f;
        for (int d = 0; d < DH; ++d) {
            float vxv = vx[vbase + (size_t)d * HWN];
            float vyv = vy[vbase + (size_t)d * HWN];
            crs = fmaf(attn[3 * PAIR_STRIDE + abase + d], vxv, crs);
            crs = fmaf(attn[1 * PAIR_STRIDE + abase + d], vyv, crs);
            slf = fmaf(attn[0 * PAIR_STRIDE + abase + d], vxv, slf);
            slf = fmaf(attn[2 * PAIR_STRIDE + abase + d], vyv, slf);
        }
        out[idx] = crs;
        out[OUT_N + idx] = slf;
    }
#endif
}

// ---------------------------------------------------------------------------
extern "C" void kernel_launch(void* const* d_in, const int* in_sizes, int n_in,
                              void* d_out, int out_size, void* d_ws, size_t ws_size,
                              hipStream_t stream) {
    (void)in_sizes; (void)n_in; (void)out_size; (void)ws_size;
    const float* x     = (const float*)d_in[0];
    const float* y     = (const float*)d_in[1];
    const float* w_qkv = (const float*)d_in[2];
    const float* w_dw  = (const float*)d_in[3];
    const float* temp  = (const float*)d_in[4];

    char* ws = (char*)d_ws;
    size_t off = 0;
    auto alloc = [&](size_t bytes) -> void* {
        void* p = ws + off;
        off = (off + bytes + 255) & ~(size_t)255;
        return p;
    };
    const size_t T_BYTES  = (size_t)BB * C3 * HWN * sizeof(__bf16);   // 37.7 MB
    const size_t XT_BYTES = (size_t)BB * CC * HWN * sizeof(__bf16);   // 12.6 MB
    const size_t QK_BYTES = (size_t)BB * CC * HWN * sizeof(__bf16);   // 12.6 MB
    const size_t V_BYTES  = (size_t)BB * CC * HWN * sizeof(float);    // 25.2 MB
    const size_t N_BYTES  = (size_t)BB * CC * sizeof(float);
    const size_t G_BYTES  = (size_t)4 * BB * HEADS * DH * DH * sizeof(float);
    const size_t W_BYTES  = (size_t)C3 * CC * sizeof(__bf16);

    __bf16* xT = (__bf16*)alloc(XT_BYTES);
    __bf16* yT = (__bf16*)alloc(XT_BYTES);
    __bf16* wbf = (__bf16*)alloc(W_BYTES);
    __bf16* tx = (__bf16*)alloc(T_BYTES);
    __bf16* ty = (__bf16*)alloc(T_BYTES);
    __bf16* qx = (__bf16*)alloc(QK_BYTES);
    __bf16* kx = (__bf16*)alloc(QK_BYTES);
    __bf16* qy = (__bf16*)alloc(QK_BYTES);
    __bf16* ky = (__bf16*)alloc(QK_BYTES);
    float*  vx = (float*)alloc(V_BYTES);
    float*  vy = (float*)alloc(V_BYTES);
    float*  invnqx = (float*)alloc(N_BYTES);
    float*  invnkx = (float*)alloc(N_BYTES);
    float*  invnqy = (float*)alloc(N_BYTES);
    float*  invnky = (float*)alloc(N_BYTES);
    float*  gram   = (float*)alloc(G_BYTES);

    // 0) one-time conversions: weights -> bf16, inputs -> transposed bf16
    cvt_bf16<<<(C3 * CC + 255) / 256, 256, 0, stream>>>(w_qkv, wbf, C3 * CC);
    transpose_cvt<<<BB * 6 * 128, 256, 0, stream>>>(x, xT);
    transpose_cvt<<<BB * 6 * 128, 256, 0, stream>>>(y, yT);
    // 1) qkv 1x1 conv (GEMM) for x and y
    qkv1x1_wmma<<<9216, 256, 0, stream>>>(xT, wbf, tx);
    qkv1x1_wmma<<<9216, 256, 0, stream>>>(yT, wbf, ty);
    // 2) depthwise 3x3 + split + inv-norms
    dw3x3_split<<<BB * C3, 256, 0, stream>>>(tx, w_dw, qx, kx, vx, invnqx, invnkx);
    dw3x3_split<<<BB * C3, 256, 0, stream>>>(ty, w_dw, qy, ky, vy, invnqy, invnky);
    // 3) raw Gram matrices (4 pairs)
    gram_wmma<<<144, 256, 0, stream>>>(qx, kx, qy, ky, gram);
    // 4) normalization folding + temperature + softmax
    scale_softmax<<<24, 256, 0, stream>>>(gram, invnqx, invnkx, invnqy, invnky, temp);
    // 5) attn @ v -> cross (first half of d_out) and self (second half)
    attn_apply<<<3072, 256, 0, stream>>>(gram, vx, vy, (float*)d_out);
}
